// MgUnet_89850715833170
// MI455X (gfx1250) — matrix operand order repaired and checked by reading, not compiled, not run on hardware
//
#include <hip/hip_runtime.h>
#include <hip/hip_bf16.h>

// MI455X / gfx1250: wave32, WMMA 16x16x32 bf16 (f32 accum).
// GEMM-dominant workload (6x [32768x1024x1024] GEMMs ~ 0.41 TFLOP) -> bf16 WMMA path.
// Tile staging uses GLOBAL_LOAD_ASYNC_TO_LDS (ASYNCcnt) when available.

typedef __attribute__((ext_vector_type(16))) __bf16 v16bf;
typedef __attribute__((ext_vector_type(8)))  float  v8f;

// Pointer element type expected by __builtin_amdgcn_global_load_async_to_lds_b128
typedef int v4i_ __attribute__((vector_size(16)));
typedef __attribute__((address_space(1))) v4i_ gv4i;   // global (HIP __device__) AS
typedef __attribute__((address_space(3))) v4i_ sv4i;   // LDS AS

union FragBF {
    unsigned u[8];
    v16bf    v;
};

#if __has_builtin(__builtin_amdgcn_global_load_async_to_lds_b128)
#define ASYNC_LDS 1
#else
#define ASYNC_LDS 0
#endif

static __device__ __forceinline__ void async_wait0() {
#if __has_builtin(__builtin_amdgcn_s_wait_asynccnt)
    __builtin_amdgcn_s_wait_asynccnt(0);
#else
    asm volatile("s_wait_asynccnt 0x0" ::: "memory");
#endif
}

static __device__ __forceinline__ unsigned short f32_to_bf16(float f) {
    union { float f; unsigned u; } v;
    v.f = f;
    unsigned r = v.u + 0x7FFFu + ((v.u >> 16) & 1u);   // round-to-nearest-even
    return (unsigned short)(r >> 16);
}

// ---------------------------------------------------------------------------
// Fold bias + inference BatchNorm into per-feature (mul, add).
__global__ __launch_bounds__(256) void bn_fold_kernel(
    const float* __restrict__ bias, const float* __restrict__ gamma,
    const float* __restrict__ beta, const float* __restrict__ rmean,
    const float* __restrict__ rvar, float* __restrict__ bnMul,
    float* __restrict__ bnAdd)
{
    int g = blockIdx.x * 256 + threadIdx.x;           // 0..1023
    float sc = gamma[g] * rsqrtf(rvar[g] + 1e-5f);
    bnMul[g] = sc;
    bnAdd[g] = (bias[g] - rmean[g]) * sc + beta[g];
}

// ---------------------------------------------------------------------------
// Masked row-softmax over 64x64 e; split into diag and off-diagonal parts.
__global__ void softmax_a_kernel(const float* __restrict__ e,
                                 const int* __restrict__ adj,
                                 float* __restrict__ Aoff,
                                 float* __restrict__ Adiag)
{
    int n = threadIdx.x;                               // 64 threads
    if (n >= 64) return;
    const float NEG = -9.0e15f;
    float mx = -3.0e38f;
    for (int m = 0; m < 64; ++m) {
        float lv = (adj[n * 64 + m] != 0) ? e[n * 64 + m] : NEG;
        mx = fmaxf(mx, lv);
    }
    float sum = 0.f;
    for (int m = 0; m < 64; ++m) {
        float lv = (adj[n * 64 + m] != 0) ? e[n * 64 + m] : NEG;
        sum += expf(lv - mx);
    }
    float inv = 1.0f / sum;
    for (int m = 0; m < 64; ++m) {
        float lv = (adj[n * 64 + m] != 0) ? e[n * 64 + m] : NEG;
        float a  = expf(lv - mx) * inv;
        if (m == n) { Adiag[n] = a; a = 0.f; }         // diag on h0, zeroed for h1 mix
        Aoff[n * 64 + m] = a;
    }
}

// ---------------------------------------------------------------------------
// Transpose + convert W [2][k][g] f32 -> Wt [2][g][k] bf16.
__global__ __launch_bounds__(256) void pack_w_kernel(
    const float* __restrict__ W, unsigned short* __restrict__ Wt)
{
    __shared__ float tile[32 * 33];
    int k0 = blockIdx.x * 32;
    int g0 = blockIdx.y * 32;
    int wsel = blockIdx.z;
    const float*    src = W  + ((size_t)wsel << 20);
    unsigned short* dst = Wt + ((size_t)wsel << 20);
    int t = threadIdx.x;
    {
        int kk = t >> 3, gg = (t & 7) * 4;
        const float4 d = *(const float4*)(src + (size_t)(k0 + kk) * 1024 + g0 + gg);
        tile[kk * 33 + gg + 0] = d.x; tile[kk * 33 + gg + 1] = d.y;
        tile[kk * 33 + gg + 2] = d.z; tile[kk * 33 + gg + 3] = d.w;
    }
    __syncthreads();
    {
        int gg = t >> 3, kk = (t & 7) * 4;
        unsigned u0 = (unsigned)f32_to_bf16(tile[(kk + 0) * 33 + gg]) |
                      ((unsigned)f32_to_bf16(tile[(kk + 1) * 33 + gg]) << 16);
        unsigned u1 = (unsigned)f32_to_bf16(tile[(kk + 2) * 33 + gg]) |
                      ((unsigned)f32_to_bf16(tile[(kk + 3) * 33 + gg]) << 16);
        *(uint2*)(dst + (size_t)(g0 + gg) * 1024 + k0 + kk) = make_uint2(u0, u1);
    }
}

// ---------------------------------------------------------------------------
// Gather-permute x into bf16 node matrix Xbf[bv][n][g].
__global__ __launch_bounds__(256) void pack_x_kernel(
    const float* __restrict__ x, const int* __restrict__ idx,
    unsigned short* __restrict__ Xbf, int mode)
{
    int n  = blockIdx.x;                               // 0..63
    int bv = blockIdx.y;                               // 0..511
    int b = bv >> 3, v = bv & 7;
    int pair = n >> 5, s = n & 31;
    int vsel = pair ? idx[v] : v;
    const float*    src = x   + (size_t)(b * 8 + vsel) * 32768;
    unsigned short* dst = Xbf + (size_t)bv * 65536 + (size_t)n * 1024;
    int g0 = threadIdx.x * 4;
    int i = g0 >> 5, j = g0 & 31;
    float f0, f1, f2, f3;
    if (mode == 1) {
        const float4 d = *(const float4*)(src + s * 1024 + g0);
        f0 = d.x; f1 = d.y; f2 = d.z; f3 = d.w;
    } else if (mode == 2) {
        const float4 d = *(const float4*)(src + i * 1024 + s * 32 + j);
        f0 = d.x; f1 = d.y; f2 = d.z; f3 = d.w;
    } else {
        const float* p = src + i * 1024 + j * 32 + s;  // stride-32 gather
        f0 = p[0]; f1 = p[32]; f2 = p[64]; f3 = p[96];
    }
    unsigned u0 = (unsigned)f32_to_bf16(f0) | ((unsigned)f32_to_bf16(f1) << 16);
    unsigned u1 = (unsigned)f32_to_bf16(f2) | ((unsigned)f32_to_bf16(f3) << 16);
    *(uint2*)(dst + g0) = make_uint2(u0, u1);
}

// ---------------------------------------------------------------------------
// Fused dual-GEMM (H0 = X*W0, H1 = X*W1) + graph mixing + BN/ReLU + pair-mean
// + permuted scatter. Block = 256 threads (8 wave32); grid = (8 col-tiles, 512 bv).
__global__ __launch_bounds__(256) void gemm_graph_kernel(
    const unsigned short* __restrict__ Xbf,   // [512][64][1024] bf16
    const unsigned short* __restrict__ Wt,    // [2][1024(g)][1024(k)] bf16
    const float* __restrict__ Aoff,           // [64][64]
    const float* __restrict__ Adiag,          // [64]
    const float* __restrict__ bnMul,          // [1024]
    const float* __restrict__ bnAdd,          // [1024]
    float* __restrict__ out,                  // [64][8][32][32][32]
    int mode, float coefHalf, int accumulate)
{
    extern __shared__ unsigned smem[];
    // dword layout:
    //  [    0, 1088)  X stage: 64 rows x 17 dwords (32 bf16 + pad)
    //  [ 1088, 5440)  W stage: 256 col-slots x 17 dwords
    //  [    0,16384)  after K loop: H0/H1 as f32 [2][64][128]
    //  [16384,20480)  A_off copy, [20480,20544) A_diag copy
    unsigned* sX = smem;
    unsigned* sW = smem + 1088;
    float*    sH = (float*)smem;
    float*    sA = (float*)(smem + 16384);
    float*    sD = (float*)(smem + 20480);

    const int tid  = threadIdx.x;
    const int bv   = blockIdx.y;
    const int col0 = blockIdx.x * 128;

    for (int i = tid; i < 4096; i += 256) sA[i] = Aoff[i];
    if (tid < 64) sD[tid] = Adiag[tid];

    const int wave  = tid >> 5;
    const int lane  = tid & 31;
    const int l     = lane & 15;
    const int hb    = lane >> 4;
    const int wsel  = wave >> 2;      // 0 -> W0/H0, 1 -> W1/H1
    const int cpair = wave & 3;       // 32-col slab within the 128-col tile

    v8f zero = {0.f, 0.f, 0.f, 0.f, 0.f, 0.f, 0.f, 0.f};
    v8f acc[4][2];
#pragma unroll
    for (int rt = 0; rt < 4; ++rt)
#pragma unroll
        for (int ct = 0; ct < 2; ++ct) acc[rt][ct] = zero;

    const unsigned short* Xrow = Xbf + (size_t)bv * 65536;
    const int xr = tid >> 2;          // stage row 0..63
    const int xc = tid & 3;           // 8-bf16 chunk
    const int wsel_s = tid >> 7;      // staging weight select
    const int wcol   = tid & 127;     // staging column
    const unsigned short* Wcol =
        Wt + ((size_t)wsel_s << 20) + (size_t)(col0 + wcol) * 1024;

    for (int k0 = 0; k0 < 1024; k0 += 32) {
#if ASYNC_LDS
        // Direct global -> LDS async copies (no VGPR round trip), ASYNCcnt tracked.
        __builtin_amdgcn_global_load_async_to_lds_b128(
            (gv4i*)(Xrow + xr * 1024 + k0 + xc * 8),
            (sv4i*)(sX + xr * 17 + xc * 4),
            0, 0);
        {
            const unsigned short* wsrc = Wcol + k0;
            unsigned* wdst = sW + tid * 17;
#pragma unroll
            for (int q = 0; q < 4; ++q)
                __builtin_amdgcn_global_load_async_to_lds_b128(
                    (gv4i*)(wsrc + q * 8),
                    (sv4i*)(wdst + q * 4),
                    0, 0);
        }
        async_wait0();
#else
        // stage X tile (64x32 bf16), 16B per thread
        {
            const uint4 d = *(const uint4*)(Xrow + xr * 1024 + k0 + xc * 8);
            unsigned* p = sX + xr * 17 + xc * 4;
            p[0] = d.x; p[1] = d.y; p[2] = d.z; p[3] = d.w;
        }
        // stage W tiles (2 x 32x128 bf16), 64B per thread (one column's k-run)
        {
            const uint4* src = (const uint4*)(Wcol + k0);
            unsigned* p = sW + tid * 17;
#pragma unroll
            for (int q = 0; q < 4; ++q) {
                uint4 d = src[q];
                p[q * 4 + 0] = d.x; p[q * 4 + 1] = d.y;
                p[q * 4 + 2] = d.z; p[q * 4 + 3] = d.w;
            }
        }
#endif
        __syncthreads();

        // B fragments: lanes 0-15 K=0..15, lanes 16-31 K=16..31; N = lane%16
        FragBF bfrag[2];
#pragma unroll
        for (int ct = 0; ct < 2; ++ct) {
            int colslot = (wsel << 7) + cpair * 32 + ct * 16 + l;
            unsigned* base = sW + colslot * 17 + hb * 8;
#pragma unroll
            for (int r = 0; r < 8; ++r) bfrag[ct].u[r] = base[r];
        }
        // A fragments per 16-row tile: lanes 0-15 K=0..15, lanes 16-31 K=16..31 (matches B)
#pragma unroll
        for (int rt = 0; rt < 4; ++rt) {
            FragBF afrag;
            unsigned* base = sX + (rt * 16 + l) * 17;
#pragma unroll
            for (int r = 0; r < 8; ++r) {
                int dw = hb * 8 + r;
                afrag.u[r] = base[dw];
            }
            acc[rt][0] = __builtin_amdgcn_wmma_f32_16x16x32_bf16(
                false, afrag.v, false, bfrag[0].v, (short)0, acc[rt][0], false, false);
            acc[rt][1] = __builtin_amdgcn_wmma_f32_16x16x32_bf16(
                false, afrag.v, false, bfrag[1].v, (short)0, acc[rt][1], false, false);
        }
        __syncthreads();
    }

    // Spill accumulators: C layout VGPR r -> row (hb?8:0)+r, col = lane%16
#pragma unroll
    for (int rt = 0; rt < 4; ++rt)
#pragma unroll
        for (int ct = 0; ct < 2; ++ct) {
            int col = cpair * 32 + ct * 16 + l;
#pragma unroll
            for (int r = 0; r < 8; ++r) {
                int row = rt * 16 + hb * 8 + r;
                sH[(wsel * 64 + row) * 128 + col] = acc[rt][ct][r];
            }
        }
    __syncthreads();

    // Epilogue: diag(A)*H0 + Aoff*H1 + BN + ReLU + pair-mean + scatter
    const int   g     = tid & 127;
    const int   shalf = tid >> 7;
    const int   gfull = col0 + g;
    const float m1 = bnMul[gfull];
    const float a1 = bnAdd[gfull];
    float* H0 = sH;
    float* H1 = sH + 64 * 128;
    const size_t outBase = (size_t)bv * 32768;
    const int gi = gfull >> 5, gj = gfull & 31;

#pragma unroll 1
    for (int so = 0; so < 16; ++so) {
        int s  = shalf * 16 + so;
        int n0 = s, n1 = s + 32;
        float v0 = sD[n0] * H0[n0 * 128 + g];
        float v1 = sD[n1] * H0[n1 * 128 + g];
#pragma unroll 8
        for (int m = 0; m < 64; ++m) {
            float hv = H1[m * 128 + g];
            v0 = fmaf(sA[n0 * 64 + m], hv, v0);
            v1 = fmaf(sA[n1 * 64 + m], hv, v1);
        }
        v0 = fmaf(v0, m1, a1); v0 = v0 > 0.f ? v0 : 0.f;
        v1 = fmaf(v1, m1, a1); v1 = v1 > 0.f ? v1 : 0.f;
        float y = coefHalf * (v0 + v1);
        size_t off;
        if (mode == 0)      off = (size_t)gi * 1024 + (size_t)gj * 32 + s;
        else if (mode == 1) off = (size_t)s * 1024 + gfull;
        else                off = (size_t)gi * 1024 + (size_t)s * 32 + gj;
        float* dst = out + outBase + off;
        if (accumulate) *dst += y; else *dst = y;
    }
}

// ---------------------------------------------------------------------------
extern "C" void kernel_launch(void* const* d_in, const int* in_sizes, int n_in,
                              void* d_out, int out_size, void* d_ws, size_t ws_size,
                              hipStream_t stream) {
    (void)in_sizes; (void)n_in; (void)out_size; (void)ws_size;
    const float* x   = (const float*)d_in[0];
    const int*   idx = (const int*)d_in[1];
    const int*   adj[2] = { (const int*)d_in[2], (const int*)d_in[3] };

    char* ws = (char*)d_ws;
    unsigned short* Xbf  = (unsigned short*)ws;                         // 67,108,864 B
    unsigned short* Wt   = (unsigned short*)(ws + 67108864);            //  4,194,304 B
    float* Aoff  = (float*)(ws + 67108864 + 4194304);                   //     16,384 B
    float* Adiag = Aoff + 4096;
    float* bnMul = Adiag + 64;
    float* bnAdd = bnMul + 1024;

    const float coefHalf[3] = { 0.25f, 0.125f, 0.125f };  // channel coef * pair-mean 0.5
    const int   adjSel[3]   = { 0, 1, 1 };
    const size_t SMEM = 20544u * 4u;                      // 82,176 B dynamic LDS

    for (int c = 0; c < 3; ++c) {
        const float* W     = (const float*)d_in[4 + 7 * c];
        const float* e     = (const float*)d_in[5 + 7 * c];
        const float* bias  = (const float*)d_in[6 + 7 * c];
        const float* gamma = (const float*)d_in[7 + 7 * c];
        const float* beta  = (const float*)d_in[8 + 7 * c];
        const float* rmean = (const float*)d_in[9 + 7 * c];
        const float* rvar  = (const float*)d_in[10 + 7 * c];

        bn_fold_kernel<<<dim3(4), 256, 0, stream>>>(bias, gamma, beta, rmean, rvar,
                                                    bnMul, bnAdd);
        softmax_a_kernel<<<1, 64, 0, stream>>>(e, adj[adjSel[c]], Aoff, Adiag);
        pack_w_kernel<<<dim3(32, 32, 2), 256, 0, stream>>>(W, Wt);
        pack_x_kernel<<<dim3(64, 512), 256, 0, stream>>>(x, idx, Xbf, c);
        gemm_graph_kernel<<<dim3(8, 512), 256, SMEM, stream>>>(
            Xbf, Wt, Aoff, Adiag, bnMul, bnAdd, (float*)d_out,
            c, coefHalf[c], c > 0 ? 1 : 0);
    }
}
